// LengthRegulator_44719199486178
// MI455X (gfx1250) — matrix operand order repaired and checked by reading, not compiled, not run on hardware
//
#include <hip/hip_runtime.h>

// ---------------------------------------------------------------------------
// Types for CDNA5 WMMA (wave32): V_WMMA_F32_16X16X32_BF16
// ---------------------------------------------------------------------------
typedef __attribute__((ext_vector_type(16))) __bf16 v16bf;
typedef __attribute__((ext_vector_type(8)))  float  v8f;

struct Frag32 { uint4 a, b; };

__device__ __forceinline__ v16bf load_frag(const unsigned int* p) {
  Frag32 f;
  f.a = *(const uint4*)(p);
  f.b = *(const uint4*)(p + 4);
  return __builtin_bit_cast(v16bf, f);
}

// split f32 into bf16 hi + bf16 lo (round-to-nearest-even-ish on both parts)
__device__ __forceinline__ void split_bf16(float x, unsigned short& hi, unsigned short& lo) {
  unsigned int u = __float_as_uint(x);
  unsigned int rh = (u + 0x7FFFu + ((u >> 16) & 1u)) >> 16;
  hi = (unsigned short)rh;
  float fh = __uint_as_float(rh << 16);
  float r = x - fh;
  unsigned int v = __float_as_uint(r);
  lo = (unsigned short)((v + 0x7FFFu + ((v >> 16) & 1u)) >> 16);
}

// ---------------------------------------------------------------------------
// Async copy of 64 contiguous bytes (per lane) global -> LDS via the CDNA5
// GLOBAL_LOAD_ASYNC_TO_LDS_B128 path (ASYNCcnt-tracked; INST_OFFSET is added
// to both the LDS and global addresses -> one (lds,off) pair covers 64B).
// ---------------------------------------------------------------------------
__device__ __forceinline__ void async_copy64(unsigned lds_addr, unsigned byte_off,
                                             const void* gbase) {
  asm volatile(
      "global_load_async_to_lds_b128 %0, %1, %2 offset:0\n\t"
      "global_load_async_to_lds_b128 %0, %1, %2 offset:16\n\t"
      "global_load_async_to_lds_b128 %0, %1, %2 offset:32\n\t"
      "global_load_async_to_lds_b128 %0, %1, %2 offset:48"
      :
      : "v"(lds_addr), "v"(byte_off), "s"((unsigned long long)(uintptr_t)gbase)
      : "memory");
}

__device__ __forceinline__ unsigned lds_off32(const void* p) {
  // generic pointer to a __shared__ object: low 32 bits are the LDS byte addr
  return (unsigned)(uintptr_t)p;
}

// ---------------------------------------------------------------------------
// Weight prep: w[F=256][CIN][3] f32 -> fragment-ordered bf16 hi/lo slabs.
// Global layout matches the LDS/WMMA B-fragment layout exactly:
//   dword index i = ((k*NCH + chunk)*16*32*8) + (tile*32 + lane)*8 + j
//   dword = bf16(w[f][c][k]) | bf16(w[f][c+1][k]) << 16
//   f = tile*16 + (lane&15),  c = chunk*32 + ((lane>=16)?16:0) + 2*j
// ---------------------------------------------------------------------------
template<int CIN>
__global__ void prep_w_frag(const float* __restrict__ W,
                            unsigned int* __restrict__ Fhi,
                            unsigned int* __restrict__ Flo) {
  constexpr int NCH = CIN / 32;
  const int i = blockIdx.x * 256 + threadIdx.x;
  if (i >= 3 * NCH * 4096) return;
  const int j     = i & 7;
  const int lane  = (i >> 3) & 31;
  const int tile  = (i >> 8) & 15;
  const int chunk = (i >> 12) % NCH;
  const int k     = (i >> 12) / NCH;
  const int f = tile * 16 + (lane & 15);
  const int c = chunk * 32 + ((lane >> 4) << 4) + 2 * j;
  const float e = W[((size_t)f * CIN + c) * 3 + k];
  const float o = W[((size_t)f * CIN + c + 1) * 3 + k];
  unsigned short he, le, ho, lo2;
  split_bf16(e, he, le);
  split_bf16(o, ho, lo2);
  Fhi[i] = (unsigned int)he | ((unsigned int)ho  << 16);
  Flo[i] = (unsigned int)le | ((unsigned int)lo2 << 16);
}

// ---------------------------------------------------------------------------
// Conv-as-GEMM with split-bf16 WMMA + async-DMA double-buffered B slabs.
//   X   : [16384][CIN] f32 activations (batches of 512 rows; zero pad at edges)
//   Bfr : [3*NCH][4096] uint  fragment-ordered weights (hi/lo)
//   Out : [16384][256] f32 (= conv(X,W) + bias)
// Workgroup: 256 thr (8 waves), tile 64(M) x 256(N); wave = 16(M) x 128(N).
// Per iteration: stage A (f32->split bf16, swizzled into WMMA A-frag layout),
// async-load next B chunk, s_wait_asynccnt for current chunk, barrier, 24 WMMAs.
// ---------------------------------------------------------------------------
template<int CIN>
__global__ __launch_bounds__(256) void conv_gemm(
    const float* __restrict__ X,
    const unsigned int* __restrict__ Bfrag_hi,
    const unsigned int* __restrict__ Bfrag_lo,
    const float* __restrict__ bias,
    float* __restrict__ Out)
{
  constexpr int F = 256;
  constexpr int NCH = CIN / 32;
  constexpr int NITER = 3 * NCH;

  __shared__ __align__(16) unsigned int lds_a_hi[4 * 32 * 8];
  __shared__ __align__(16) unsigned int lds_a_lo[4 * 32 * 8];
  __shared__ __align__(16) unsigned int lds_b_hi[2][16 * 32 * 8];
  __shared__ __align__(16) unsigned int lds_b_lo[2][16 * 32 * 8];

  const int tid  = threadIdx.x;
  const int row0 = blockIdx.x * 64;     // 512 % 64 == 0 -> never crosses batch
  const int l0   = row0 & 511;

  // A staging: thread -> (row m in 0..63, 8-channel group)
  const int sm  = tid >> 2;
  const int scg = (tid & 3) * 8;
  // compute mapping
  const int wv    = tid >> 5;
  const int lane  = tid & 31;
  const int mtile = wv >> 1;            // 0..3
  const int ntb   = (wv & 1) * 8;       // N-tile base (8 tiles of 16)

  // LDS byte addresses for the async engine
  const unsigned bh_base[2] = { lds_off32(&lds_b_hi[0][0]) + (unsigned)tid * 64u,
                                lds_off32(&lds_b_hi[1][0]) + (unsigned)tid * 64u };
  const unsigned bl_base[2] = { lds_off32(&lds_b_lo[0][0]) + (unsigned)tid * 64u,
                                lds_off32(&lds_b_lo[1][0]) + (unsigned)tid * 64u };

  v8f acc[8] = {};

  // prologue: DMA chunk 0 into buffer 0 (8 async ops per wave in flight)
  {
    const unsigned off0 = (unsigned)tid * 64u;
    async_copy64(bh_base[0], off0, Bfrag_hi);
    async_copy64(bl_base[0], off0, Bfrag_lo);
  }

  for (int it = 0; it < NITER; ++it) {
    __syncthreads();   // all waves done with the buffer we are about to reuse

    // ---------------- stage A (64 x 32, f32 -> split bf16, swizzled) -------
    {
      const int kk = it / NCH;
      const int cb = (it - kk * NCH) * 32;
      const int lsrc = l0 + sm + kk - 1;
      float v[8];
      if (lsrc >= 0 && lsrc < 512) {
        const float* src = X + (size_t)(row0 + sm + kk - 1) * CIN + cb + scg;
        float4 p0 = *(const float4*)(src);
        float4 p1 = *(const float4*)(src + 4);
        v[0]=p0.x; v[1]=p0.y; v[2]=p0.z; v[3]=p0.w;
        v[4]=p1.x; v[5]=p1.y; v[6]=p1.z; v[7]=p1.w;
      } else {
        #pragma unroll
        for (int i = 0; i < 8; ++i) v[i] = 0.f;
      }
      #pragma unroll
      for (int i = 0; i < 4; ++i) {
        const int c = scg + 2 * i;                 // even K of the pair
        unsigned short he, le, ho, lo2;
        split_bf16(v[2*i],   he, le);
        split_bf16(v[2*i+1], ho, lo2);
        const int ln2 = (sm & 15) + (((c >> 4) & 1) << 4);
        const int jj  = (c & 15) >> 1;
        const int dw  = (((sm >> 4) * 32 + ln2) << 3) + jj;
        lds_a_hi[dw] = (unsigned int)he | ((unsigned int)ho  << 16);
        lds_a_lo[dw] = (unsigned int)le | ((unsigned int)lo2 << 16);
      }
    }

    // ---------------- DMA next B chunk; wait for current one ---------------
    if (it + 1 < NITER) {
      const int nb = (it + 1) & 1;
      const unsigned goff = (unsigned)(it + 1) * 16384u + (unsigned)tid * 64u;
      async_copy64(bh_base[nb], goff, Bfrag_hi);
      async_copy64(bl_base[nb], goff, Bfrag_lo);
      // 8 newest ops outstanding; in-order completion => chunk `it` is in LDS
      asm volatile("s_wait_asynccnt 0x8" ::: "memory");
    } else {
      asm volatile("s_wait_asynccnt 0x0" ::: "memory");
    }
    __syncthreads();

    // ---------------- compute: 8 N-tiles x 3 split-WMMAs -------------------
    const int cbuf = it & 1;
    const v16bf ah = load_frag(&lds_a_hi[(mtile * 32 + lane) * 8]);
    const v16bf al = load_frag(&lds_a_lo[(mtile * 32 + lane) * 8]);
    #pragma unroll
    for (int t = 0; t < 8; ++t) {
      const v16bf bh = load_frag(&lds_b_hi[cbuf][((ntb + t) * 32 + lane) * 8]);
      const v16bf bl = load_frag(&lds_b_lo[cbuf][((ntb + t) * 32 + lane) * 8]);
      acc[t] = __builtin_amdgcn_wmma_f32_16x16x32_bf16(false, ah, false, bh, (short)0, acc[t], false, false);
      acc[t] = __builtin_amdgcn_wmma_f32_16x16x32_bf16(false, ah, false, bl, (short)0, acc[t], false, false);
      acc[t] = __builtin_amdgcn_wmma_f32_16x16x32_bf16(false, al, false, bh, (short)0, acc[t], false, false);
    }
  }

  // ---------------- epilogue: C layout -> Out + bias -----------------------
  const int rowb = row0 + mtile * 16 + ((lane >> 4) << 3);
  #pragma unroll
  for (int t = 0; t < 8; ++t) {
    const int col = (ntb + t) * 16 + (lane & 15);
    const float bc = bias[col];
    #pragma unroll
    for (int r = 0; r < 8; ++r) {
      Out[(size_t)(rowb + r) * F + col] = acc[t][r] + bc;
    }
  }
}

// ---------------------------------------------------------------------------
// LayerNorm(F=256) + ReLU -> f32 (input for conv2)
// ---------------------------------------------------------------------------
__global__ __launch_bounds__(256) void ln_relu(
    const float* __restrict__ H, const float* __restrict__ g,
    const float* __restrict__ b, float* __restrict__ Outp)
{
  __shared__ float red[256];
  const int row = blockIdx.x, f = threadIdx.x;
  const float v = H[(size_t)row * 256 + f];
  red[f] = v; __syncthreads();
  for (int s = 128; s > 0; s >>= 1) { if (f < s) red[f] += red[f + s]; __syncthreads(); }
  const float mean = red[0] * (1.f / 256.f);
  __syncthreads();
  const float d = v - mean;
  red[f] = d * d; __syncthreads();
  for (int s = 128; s > 0; s >>= 1) { if (f < s) red[f] += red[f + s]; __syncthreads(); }
  const float var = red[0] * (1.f / 256.f);
  const float y = d * rsqrtf(var + 1e-5f) * g[f] + b[f];
  Outp[(size_t)row * 256 + f] = fmaxf(y, 0.f);
}

// ---------------------------------------------------------------------------
// LayerNorm + ReLU + dot(lin_w) + lin_b -> duration[row]
// ---------------------------------------------------------------------------
__global__ __launch_bounds__(256) void ln_dot(
    const float* __restrict__ H, const float* __restrict__ g,
    const float* __restrict__ b, const float* __restrict__ lw,
    const float* __restrict__ lb, float* __restrict__ dur)
{
  __shared__ float red[256];
  const int row = blockIdx.x, f = threadIdx.x;
  const float v = H[(size_t)row * 256 + f];
  red[f] = v; __syncthreads();
  for (int s = 128; s > 0; s >>= 1) { if (f < s) red[f] += red[f + s]; __syncthreads(); }
  const float mean = red[0] * (1.f / 256.f);
  __syncthreads();
  const float d = v - mean;
  red[f] = d * d; __syncthreads();
  for (int s = 128; s > 0; s >>= 1) { if (f < s) red[f] += red[f + s]; __syncthreads(); }
  const float var = red[0] * (1.f / 256.f);
  __syncthreads();
  const float y = fmaxf(d * rsqrtf(var + 1e-5f) * g[f] + b[f], 0.f);
  red[f] = y * lw[f]; __syncthreads();
  for (int s = 128; s > 0; s >>= 1) { if (f < s) red[f] += red[f + s]; __syncthreads(); }
  if (f == 0) dur[row] = red[0] + lb[0];
}

// ---------------------------------------------------------------------------
// Inclusive cumsum of target along L=512 (one block per batch row)
// ---------------------------------------------------------------------------
__global__ void cumsum_rows(const int* __restrict__ tgt, int* __restrict__ cums) {
  __shared__ int s[512];
  const int n = blockIdx.x, l = threadIdx.x;
  s[l] = tgt[n * 512 + l];
  __syncthreads();
  for (int off = 1; off < 512; off <<= 1) {
    const int v = (l >= off) ? s[l - off] : 0;
    __syncthreads();
    s[l] += v;
    __syncthreads();
  }
  cums[n * 512 + l] = s[l];
}

// ---------------------------------------------------------------------------
// Length regulator: out[n,t,:] = x[n, searchsorted(cums[n], t, right), :]
// (zero if t >= total). One block per (n, t). Pure-bandwidth stage.
// ---------------------------------------------------------------------------
__global__ __launch_bounds__(128) void length_reg(
    const float* __restrict__ X, const int* __restrict__ cums,
    float* __restrict__ Outp)
{
  const int bid = blockIdx.x;
  const int n = bid >> 11;           // T = 2048
  const int t = bid & 2047;
  const int* cr = cums + n * 512;
  const int total = cr[511];
  int lo = 0, hi = 512;
  while (lo < hi) { const int mid = (lo + hi) >> 1; if (cr[mid] <= t) lo = mid + 1; else hi = mid; }
  const int idx = lo < 512 ? lo : 511;
  const bool valid = t < total;
  const float* src = X + ((size_t)n * 512 + idx) * 384;
  float* dst = Outp + ((size_t)n * 2048 + t) * 384;
  __builtin_prefetch(&src[threadIdx.x], 0, 0);   // global_prefetch_b8
  for (int d = threadIdx.x; d < 384; d += 128)
    dst[d] = valid ? src[d] : 0.f;
}

// ---------------------------------------------------------------------------
extern "C" void kernel_launch(void* const* d_in, const int* in_sizes, int n_in,
                              void* d_out, int out_size, void* d_ws, size_t ws_size,
                              hipStream_t stream) {
  const float* x   = (const float*)d_in[0];   // [32,512,384]
  const float* w1  = (const float*)d_in[1];   // [256,384,3]
  const float* b1  = (const float*)d_in[2];
  const float* g1  = (const float*)d_in[3];
  const float* be1 = (const float*)d_in[4];
  const float* w2  = (const float*)d_in[5];   // [256,256,3]
  const float* b2  = (const float*)d_in[6];
  const float* g2  = (const float*)d_in[7];
  const float* be2 = (const float*)d_in[8];
  const float* lw  = (const float*)d_in[9];
  const float* lb  = (const float*)d_in[10];
  const int*   tgt = (const int*)d_in[11];
  // d_in[12] = mel_max_length (2048, hardcoded)

  float* out = (float*)d_out;                       // [32,2048,384]
  float* dur = out + (size_t)32 * 2048 * 384;       // [32,512]

  char* ws = (char*)d_ws;
  size_t off = 0;
  auto walloc = [&](size_t bytes) { void* p = ws + off; off += (bytes + 255) & ~(size_t)255; return p; };
  unsigned int* w1f_hi = (unsigned int*)walloc((size_t)3 * 12 * 4096 * 4);
  unsigned int* w1f_lo = (unsigned int*)walloc((size_t)3 * 12 * 4096 * 4);
  unsigned int* w2f_hi = (unsigned int*)walloc((size_t)3 * 8 * 4096 * 4);
  unsigned int* w2f_lo = (unsigned int*)walloc((size_t)3 * 8 * 4096 * 4);
  float* h1  = (float*)walloc((size_t)16384 * 256 * 4);
  float* h1n = (float*)walloc((size_t)16384 * 256 * 4);
  float* h2  = (float*)walloc((size_t)16384 * 256 * 4);
  int*   cums = (int*)walloc((size_t)32 * 512 * 4);
  (void)ws_size; (void)in_sizes; (void)n_in; (void)out_size;

  prep_w_frag<384><<<(3 * 12 * 4096) / 256, 256, 0, stream>>>(w1, w1f_hi, w1f_lo);
  prep_w_frag<256><<<(3 * 8 * 4096) / 256, 256, 0, stream>>>(w2, w2f_hi, w2f_lo);

  conv_gemm<384><<<256, 256, 0, stream>>>(x, w1f_hi, w1f_lo, b1, h1);
  ln_relu<<<16384, 256, 0, stream>>>(h1, g1, be1, h1n);
  conv_gemm<256><<<256, 256, 0, stream>>>(h1n, w2f_hi, w2f_lo, b2, h2);
  ln_dot<<<16384, 256, 0, stream>>>(h2, g2, be2, lw, lb, dur);

  cumsum_rows<<<32, 512, 0, stream>>>(tgt, cums);
  length_reg<<<32 * 2048, 128, 0, stream>>>(x, cums, out);
}
